// MultiTowerEdge_83356725281388
// MI455X (gfx1250) — matrix operand order repaired and checked by reading, not compiled, not run on hardware
//
#include <hip/hip_runtime.h>

// Problem constants (match reference)
#define N_EDGES    200000
#define N_NODES    20000
#define D_NODE     128
#define D_EDGE     32
#define EPT        16                    // edges per tile (WMMA M/N = 16)
#define N_TILES    (N_EDGES / EPT)       // 12500
#define NBR_STRIDE 132                   // 128 + 4 pad -> conflict-free LDS rows
#define OUT_HALF   (EPT * NBR_STRIDE + 32)   // 2144 words; %64==32 -> halves hit disjoint banks

typedef __bf16 bf16_t;
typedef __attribute__((ext_vector_type(16))) __bf16 v16bf;
typedef __attribute__((ext_vector_type(8)))  float  v8f;
typedef __attribute__((ext_vector_type(4)))  float  fvec4;   // native clang vector (not HIP struct)
typedef int v4i_ __attribute__((vector_size(16)));           // v4i32 for async-LDS builtin
typedef __attribute__((address_space(1))) v4i_* gv4i_p;      // global (AS1)
typedef __attribute__((address_space(3))) v4i_* lv4i_p;      // LDS (AS3)

#if __has_builtin(__builtin_amdgcn_global_load_async_to_lds_b128)
#define HAVE_ASYNC_LDS 1
#else
#define HAVE_ASYNC_LDS 0
#endif

#if __has_builtin(__builtin_amdgcn_sched_barrier)
// allow VMEM-read (0x20) and DS-read (0x100) to cross; pin WMMA/VALU to cap
// accumulator liveness (keeps allocation under 256 VGPRs -> no s_set_vgpr_msb)
#define SCHED_FENCE() __builtin_amdgcn_sched_barrier(0x120)
#else
#define SCHED_FENCE() asm volatile("" ::: "memory")
#endif

// Single-instruction relu that stays compiler-visible (hazard-safe):
// med3(x, 0, FLT_MAX) == min(max(x,0), FLT_MAX) == relu for finite x.
// (A +inf bound gets instcombined back to maxnum -> canonicalize+max;
//  inline asm is invisible to the WMMA->VALU hazard recognizer.)
static __device__ __forceinline__ float relu1(float x) {
#if __has_builtin(__builtin_amdgcn_fmed3f)
    return __builtin_amdgcn_fmed3f(x, 0.0f, 3.402823466e+38f);
#else
    return fmaxf(x, 0.0f);
#endif
}

__global__ __launch_bounds__(256)
void mte_wmma_kernel(const float* __restrict__ nodef,   // [N_NODES, 128]
                     const float* __restrict__ ef,      // [N_EDGES, 32]
                     const int*   __restrict__ eidx,    // [2, N_EDGES]
                     const float* __restrict__ W,       // [2048, 32]
                     const float* __restrict__ bias,    // [2048]
                     float*       __restrict__ out)     // [N_EDGES, 128]
{
    __shared__ float s_bias[D_NODE * 16];          // 2048 f32
    __shared__ float s_nbr[EPT * NBR_STRIDE];      // gathered neighbor rows
    __shared__ float s_out[2 * OUT_HALF];          // per-lane-half partial sums

    const int t     = threadIdx.x;
    const int wv    = t >> 5;        // wave id 0..7 == tower id
    const int ln    = t & 31;        // lane in wave32
    const int l15   = ln & 15;
    const int lhalf = ln >> 4;       // 0 / 1

    // ---- stage bias into LDS (once) ----
    for (int i = t * 4; i < D_NODE * 16; i += 256 * 4)
        *(fvec4*)&s_bias[i] = *(const fvec4*)&bias[i];

    // ---- build loop-invariant A operands: A_i[m,c] = W[256*wv + 16*i + m, c] ----
    // bf16 A 16x32 layout: lane m holds row m; lanes<16: K=0..7,16..23;
    // lanes>=16: K=8..15,24..31 (two 8-wide chunks at offsets 8*lhalf, 16+8*lhalf).
    v16bf Areg[16];
#pragma unroll
    for (int i = 0; i < 16; ++i) {
        const int row = 256 * wv + 16 * i + l15;
        const float* wr = W + row * D_EDGE + 8 * lhalf;
        fvec4 w0 = *(const fvec4*)(wr);
        fvec4 w1 = *(const fvec4*)(wr + 4);
        fvec4 w2 = *(const fvec4*)(wr + 16);
        fvec4 w3 = *(const fvec4*)(wr + 20);
        float lo[8] = {w0.x, w0.y, w0.z, w0.w, w1.x, w1.y, w1.z, w1.w};
        float hi[8] = {w2.x, w2.y, w2.z, w2.w, w3.x, w3.y, w3.z, w3.w};
        v16bf a;
#pragma unroll
        for (int q = 0; q < 8; ++q) {
            a[q]     = (bf16_t)lo[q];
            a[8 + q] = (bf16_t)hi[q];
        }
        Areg[i] = a;
    }
    __syncthreads();

    // ---- persistent grid-stride loop over edge tiles ----
    for (int tile = blockIdx.x; tile < N_TILES; tile += gridDim.x) {
        const int e0 = tile * EPT;

        __syncthreads();          // protect s_nbr/s_out reuse across iterations

        // cooperative neighbor gather: s_nbr[e][0..127] = nodef[idx[e]][:]
        // (memory -> LDS without round-tripping VGPRs; tracked by ASYNCcnt)
        {
            const int ge = t >> 4;
            const int gc = (t & 15) * 8;
            const int node = eidx[N_EDGES + e0 + ge];   // 16-lane dup -> coalesced, L2-hot
            const float* src = nodef + (long)node * D_NODE + gc;
            float* dst = &s_nbr[ge * NBR_STRIDE + gc];
#if HAVE_ASYNC_LDS
            __builtin_amdgcn_global_load_async_to_lds_b128(
                (gv4i_p)(src),     (lv4i_p)(dst),     0, 0);
            __builtin_amdgcn_global_load_async_to_lds_b128(
                (gv4i_p)(src + 4), (lv4i_p)(dst + 4), 0, 0);
#if __has_builtin(__builtin_amdgcn_s_wait_asynccnt)
            __builtin_amdgcn_s_wait_asynccnt(0);
#else
            asm volatile("s_wait_asynccnt 0x0" ::: "memory");
#endif
#else
            fvec4 a0 = *(const fvec4*)(src);
            fvec4 a1 = *(const fvec4*)(src + 4);
            *(fvec4*)(dst)     = a0;
            *(fvec4*)(dst + 4) = a1;
#endif
        }
        __syncthreads();

        // B operand: B[c, e] = ef[e0+e, c]; lane holds col e=l15, K = 8*lhalf..+7, 16+8*lhalf..+7
        // ef is streamed once -> nontemporal loads
        const fvec4* efr = (const fvec4*)(ef + (long)(e0 + l15) * D_EDGE + 8 * lhalf);
        fvec4 c0 = __builtin_nontemporal_load(efr);
        fvec4 c1 = __builtin_nontemporal_load(efr + 1);
        fvec4 c2 = __builtin_nontemporal_load(efr + 4);
        fvec4 c3 = __builtin_nontemporal_load(efr + 5);
        float cf[16] = {c0.x, c0.y, c0.z, c0.w, c1.x, c1.y, c1.z, c1.w,
                        c2.x, c2.y, c2.z, c2.w, c3.x, c3.y, c3.z, c3.w};
        v16bf Bv;
#pragma unroll
        for (int q = 0; q < 16; ++q) Bv[q] = (bf16_t)cf[q];

        // neighbor values this lane needs for the tower-wv matvec:
        // nbr2[r] = nbr[e=l15][16*wv + r + 8*lhalf]  (8 consecutive -> 2x b128)
        float nbr2[8];
        {
            const float* np = &s_nbr[l15 * NBR_STRIDE + 16 * wv + 8 * lhalf];
            fvec4 n0 = *(const fvec4*)(np);
            fvec4 n1 = *(const fvec4*)(np + 4);
            nbr2[0] = n0.x; nbr2[1] = n0.y; nbr2[2] = n0.z; nbr2[3] = n0.w;
            nbr2[4] = n1.x; nbr2[5] = n1.y; nbr2[6] = n1.z; nbr2[7] = n1.w;
        }

        // bias loader: bias is a function of M only -> it IS the WMMA C operand
        auto load_biasC = [&](int i) -> v8f {
            const float* bb = &s_bias[256 * wv + 16 * i + 8 * lhalf];
            fvec4 b0 = *(const fvec4*)(bb);
            fvec4 b1 = *(const fvec4*)(bb + 4);
            v8f c;
            c[0] = b0.x; c[1] = b0.y; c[2] = b0.z; c[3] = b0.w;
            c[4] = b1.x; c[5] = b1.y; c[6] = b1.z; c[7] = b1.w;
            return c;
        };

        // 16 output tiles (tile i == matrix-row block i), in groups of 4:
        // 4 in-flight WMMAs cover the WMMA->VALU hazard window while keeping
        // only 32 accumulator VGPRs live (total stays under 256).
#pragma unroll
        for (int g = 0; g < 4; ++g) {
            v8f acc[4];
#pragma unroll
            for (int u = 0; u < 4; ++u) {
                const int i = 4 * g + u;
                acc[u] = __builtin_amdgcn_wmma_f32_16x16x32_bf16(
                    false, Areg[i], false, Bv, (short)0, load_biasC(i),
                    false, false);
            }
#pragma unroll
            for (int u = 0; u < 4; ++u) {
                const int i = 4 * g + u;
                float p = 0.0f;
#pragma unroll
                for (int r = 0; r < 8; ++r)
                    p = fmaf(relu1(acc[u][r]), nbr2[r], p);
                // store this lane-half's partial; writeback adds the halves.
                s_out[lhalf * OUT_HALF + l15 * NBR_STRIDE + 16 * wv + i] = p;
            }
            SCHED_FENCE();
        }
        __syncthreads();

        // coalesced writeback of the 16x128 output tile (sum of both halves)
        {
            const int ge = t >> 4;
            const int gc = (t & 15) * 8;
            const int o  = ge * NBR_STRIDE + gc;
            fvec4 a0 = *(const fvec4*)&s_out[o];
            fvec4 a1 = *(const fvec4*)&s_out[o + 4];
            fvec4 b0 = *(const fvec4*)&s_out[OUT_HALF + o];
            fvec4 b1 = *(const fvec4*)&s_out[OUT_HALF + o + 4];
            fvec4 r0 = a0 + b0;
            fvec4 r1 = a1 + b1;
            fvec4* dst = (fvec4*)(out + (long)(e0 + ge) * D_NODE + gc);
            __builtin_nontemporal_store(r0, dst);       // write-once stream
            __builtin_nontemporal_store(r1, dst + 1);
        }

        // warm next tile's edge features in L2 (speculative prefetch)
        {
            const int nexte = e0 + (int)gridDim.x * EPT;
            if (nexte < N_EDGES)
                __builtin_prefetch(ef + (long)nexte * D_EDGE, 0, 0);
        }
    }
}

extern "C" void kernel_launch(void* const* d_in, const int* in_sizes, int n_in,
                              void* d_out, int out_size, void* d_ws, size_t ws_size,
                              hipStream_t stream) {
    const float* nodef = (const float*)d_in[0];   // node_features [20000,128] f32
    const float* ef    = (const float*)d_in[1];   // edge_features [200000,32] f32
    const int*   eidx  = (const int*)d_in[2];     // edge_index [2,200000] int
    const float* W     = (const float*)d_in[3];   // [2048,32] f32
    const float* bias  = (const float*)d_in[4];   // [2048] f32
    float* out = (float*)d_out;                   // [200000,128] f32

    (void)in_sizes; (void)n_in; (void)out_size; (void)d_ws; (void)ws_size;

    dim3 grid(2048), block(256);
    mte_wmma_kernel<<<grid, block, 0, stream>>>(nodef, ef, eidx, W, bias, out);
}
